// FlowProcessor_21560735826506
// MI455X (gfx1250) — matrix operand order repaired and checked by reading, not compiled, not run on hardware
//
#include <hip/hip_runtime.h>
#include <math.h>

// ---------------------------------------------------------------------------
// FlowProcessor on MI455X (gfx1250, wave32, WMMA)
//   phase 1: energy GEMM (per-flow 768-dot, coalesced, L2-resident weights)
//   phase 2: 24 steps of batched tiny-MLP via v_wmma_f32_16x16x32_bf16
//   - per-wave LDS buffers, wave-local s_wait_dscnt fences (no block barriers
//     in the hot loop; DS ops from a wave complete in order per CDNA5 ISA)
//   - dead flows free-run (owner-side `active` register gates all deposits,
//     so un-gated h/pos of dead flows are unobservable -> no afac traffic)
// ---------------------------------------------------------------------------

typedef __attribute__((ext_vector_type(16))) __bf16        v16bf;
typedef __attribute__((ext_vector_type(8)))  float         v8f;
typedef __attribute__((ext_vector_type(4)))  unsigned int  u32x4;
typedef __attribute__((ext_vector_type(8)))  unsigned int  u32x8;

#define WAVES 4
#define TPB   (WAVES * 32)

__device__ __forceinline__ unsigned short f2bf(float f) {
  unsigned u = __float_as_uint(f);
  u += 0x7FFFu + ((u >> 16) & 1u);          // round-to-nearest-even
  return (unsigned short)(u >> 16);
}
__device__ __forceinline__ float bf2f(unsigned short h) {
  return __uint_as_float(((unsigned)h) << 16);
}
__device__ __forceinline__ float fast_tanh(float x) {
#if __has_builtin(__builtin_amdgcn_tanhf)
  return __builtin_amdgcn_tanhf(x);          // V_TANH_F32 on gfx1250
#elif __has_builtin(__builtin_amdgcn_tanh_f32)
  return __builtin_amdgcn_tanh_f32(x);
#else
  return tanhf(x);
#endif
}

// Wave-local LDS producer->consumer fence. DS instructions from one wave are
// processed in order by the LDS pipe (CDNA5 ISA ch.5 "LDS: in order"); the
// s_wait_dscnt plus the compiler memory barrier makes cross-lane RAW through
// this wave's private LDS buffers safe without a workgroup barrier.
__device__ __forceinline__ void wave_lds_fence() {
  asm volatile("s_wait_dscnt 0" ::: "memory");
}

// A-matrix fragment (16x32 bf16): lane(l%16)=row M, K = {8h..8h+7, 16+8h..+7}
__device__ __forceinline__ v16bf load_fragA(const unsigned short* p) {
  u32x4 q0 = *(const u32x4*)p;          // K[0..7]   of this lane-half
  u32x4 q1 = *(const u32x4*)(p + 16);   // K[16..23] of this lane-half
  u32x8 r;
  r[0]=q0[0]; r[1]=q0[1]; r[2]=q0[2]; r[3]=q0[3];
  r[4]=q1[0]; r[5]=q1[1]; r[6]=q1[2]; r[7]=q1[3];
  return __builtin_bit_cast(v16bf, r);
}
// B-matrix fragment (32x16 bf16): lane(l%16)=col N, K = {8h..8h+7, 16+8h..+7}
__device__ __forceinline__ v16bf load_fragB(const unsigned short* p) {
  u32x4 q0 = *(const u32x4*)p;
  u32x4 q1 = *(const u32x4*)(p + 16);
  u32x8 r;
  r[0]=q0[0]; r[1]=q0[1]; r[2]=q0[2]; r[3]=q0[3];
  r[4]=q1[0]; r[5]=q1[1]; r[6]=q1[2]; r[7]=q1[3];
  return __builtin_bit_cast(v16bf, r);
}

__global__ void zero_f32(float* __restrict__ p, int n) {
  int i = blockIdx.x * blockDim.x + threadIdx.x;
  if (i < n) p[i] = 0.0f;
}

__global__ __launch_bounds__(TPB) void flow_kernel(
    const float* __restrict__ emb,   const float* __restrict__ Wmap,
    const float* __restrict__ bmap,  const float* __restrict__ winit,
    const float* __restrict__ Wpos,  const float* __restrict__ W1,
    const float* __restrict__ b1g,   const float* __restrict__ W2,
    const float* __restrict__ b2g,   const float* __restrict__ woutg,
    float* __restrict__ out)
{
  // ---- LDS: shared bf16 weights (transposed, zero padded) + per-wave buffers
  __shared__ __align__(16) unsigned short sW1T[64 * 64];   // [n][k], k 35->64 pad
  __shared__ __align__(16) unsigned short sW2T[48 * 64];   // [n][k], n 35->48 pad
  __shared__ float sB1[64], sB2[48], sWout[32], sWi[32], sWp[96];
  __shared__ __align__(16) unsigned short sFeat[WAVES][32 * 64]; // [flow][h,pn,0]
  __shared__ __align__(16) unsigned short sHid [WAVES][32 * 64]; // [flow][hidden]
  __shared__ float sDxyz[WAVES][32 * 4];

  const int tid  = threadIdx.x;
  const int w    = tid >> 5;
  const int lane = tid & 31;
  const int hf   = (lane >> 4) & 1;     // lane half
  const int lrow = lane & 15;

  // ---- stage weights into LDS (bf16, transposed) ----
  for (int i = tid; i < 64 * 64; i += TPB) {
    int n = i >> 6, k = i & 63;
    sW1T[i] = (k < 35) ? f2bf(W1[k * 64 + n]) : (unsigned short)0;
  }
  for (int i = tid; i < 48 * 64; i += TPB) {
    int n = i >> 6, k = i & 63;
    sW2T[i] = (n < 35) ? f2bf(W2[k * 35 + n]) : (unsigned short)0;
  }
  if (tid < 64) sB1[tid] = b1g[tid];
  if (tid < 48) sB2[tid] = (tid < 35) ? b2g[tid] : 0.0f;
  if (tid < 32) { sWout[tid] = woutg[tid]; sWi[tid] = winit[tid]; }
  if (tid < 96) sWp[tid] = Wpos[tid];
  __syncthreads();   // weights visible to all waves (only block barrier needed)

  // ---- per-flow init; this thread owns flow (blockIdx.x*TPB + tid) ----
  const int   batch = blockIdx.x >> 5;                 // 32 blocks per batch item
  const int   cell  = ((blockIdx.x & 31) << 7) + tid;  // 0..4095
  const float x0 = (float)(cell & 63), y0 = (float)(cell >> 6);

  float e = bmap[cell];
  {
    const float* er = emb + batch * 768;               // uniform per block
    const float* wc = Wmap + cell;                     // coalesced across lanes
    #pragma unroll 4
    for (int k = 0; k < 768; ++k) {
      if ((k & 63) == 0) __builtin_prefetch(&wc[(size_t)(k + 64) << 12], 0, 1);
      e = fmaf(er[k], wc[(size_t)k << 12], e);
    }
  }

  float px = x0, py = y0, pz = 0.0f;
  bool  active = true;
  unsigned short* frow = &sFeat[w][lane * 64];
  {
    const float pnx = x0 * (1.0f / 64.0f), pny = y0 * (1.0f / 64.0f);
    #pragma unroll
    for (int d = 0; d < 32; ++d)
      frow[d] = f2bf(fast_tanh(fmaf(e, sWi[d], fmaf(pnx, sWp[d], pny * sWp[32 + d]))));
    frow[32] = f2bf(pnx); frow[33] = f2bf(pny); frow[34] = 0;
    #pragma unroll
    for (int d = 35; d < 64; ++d) frow[d] = 0;       // K padding stays zero
  }
  wave_lds_fence();

  // per-lane bias values in C/D layout (N = 16t + lane%16)
  float b1v[4], b2v[3];
  #pragma unroll
  for (int t = 0; t < 4; ++t) b1v[t] = sB1[16 * t + lrow];
  #pragma unroll
  for (int t = 0; t < 3; ++t) b2v[t] = sB2[16 * t + lrow];

  // hidden state h in registers, distributed in GEMM2 D-layout (t=0,1 -> dims 0..31)
  float hreg[2][2][8];
  #pragma unroll
  for (int m = 0; m < 2; ++m)
    #pragma unroll
    for (int t = 0; t < 2; ++t)
      #pragma unroll
      for (int j = 0; j < 8; ++j)
        hreg[m][t][j] = bf2f(sFeat[w][(16 * m + j + 8 * hf) * 64 + 16 * t + lrow]);

  const int kA = 8 * hf;        // A-fragment K sub-offset per lane half
  const int kB = 8 * hf;        // B-fragment K sub-offset per lane half
  float* outSurf = out + batch * 4096;

  #pragma unroll 1
  for (int step = 0; step < 24; ++step) {
    // ---- GEMM1: y1 = feat @ W1 + b1  (M=32 in 2 tiles, N=64 in 4, K=64 in 2)
    v16bf a1[2][2];
    #pragma unroll
    for (int m = 0; m < 2; ++m)
      #pragma unroll
      for (int c = 0; c < 2; ++c)
        a1[m][c] = load_fragA(&sFeat[w][(16 * m + lrow) * 64 + 32 * c + kA]);

    #pragma unroll
    for (int t = 0; t < 4; ++t) {
      v16bf bA = load_fragB(&sW1T[(16 * t + lrow) * 64 +  0 + kB]);
      v16bf bB = load_fragB(&sW1T[(16 * t + lrow) * 64 + 32 + kB]);
      #pragma unroll
      for (int m = 0; m < 2; ++m) {
        v8f acc;
        #pragma unroll
        for (int j = 0; j < 8; ++j) acc[j] = b1v[t];
        acc = __builtin_amdgcn_wmma_f32_16x16x32_bf16(false, a1[m][0], false, bA, (short)0, acc, false, false);
        acc = __builtin_amdgcn_wmma_f32_16x16x32_bf16(false, a1[m][1], false, bB, (short)0, acc, false, false);
        // hidden = tanh(y1) -> LDS bf16 [flow][64]
        #pragma unroll
        for (int j = 0; j < 8; ++j)
          sHid[w][(16 * m + j + 8 * hf) * 64 + 16 * t + lrow] = f2bf(fast_tanh(acc[j]));
      }
    }
    wave_lds_fence();

    // ---- GEMM2: y2 = hidden @ W2 + b2  (N=48 in 3 tiles, K=64 in 2) ----
    v16bf a2[2][2];
    #pragma unroll
    for (int m = 0; m < 2; ++m)
      #pragma unroll
      for (int c = 0; c < 2; ++c)
        a2[m][c] = load_fragA(&sHid[w][(16 * m + lrow) * 64 + 32 * c + kA]);

    #pragma unroll
    for (int t = 0; t < 3; ++t) {
      v16bf bA = load_fragB(&sW2T[(16 * t + lrow) * 64 +  0 + kB]);
      v16bf bB = load_fragB(&sW2T[(16 * t + lrow) * 64 + 32 + kB]);
      #pragma unroll
      for (int m = 0; m < 2; ++m) {
        v8f acc;
        #pragma unroll
        for (int j = 0; j < 8; ++j) acc[j] = b2v[t];
        acc = __builtin_amdgcn_wmma_f32_16x16x32_bf16(false, a2[m][0], false, bA, (short)0, acc, false, false);
        acc = __builtin_amdgcn_wmma_f32_16x16x32_bf16(false, a2[m][1], false, bB, (short)0, acc, false, false);
        if (t < 2) {
          // h += tanh(y2[:, :32])  (lane-local), write updated h -> feat
          #pragma unroll
          for (int j = 0; j < 8; ++j) {
            float v = hreg[m][t][j] + fast_tanh(acc[j]);
            hreg[m][t][j] = v;
            sFeat[w][(16 * m + j + 8 * hf) * 64 + 16 * t + lrow] = f2bf(v);
          }
        } else if (lrow < 3) {
          // raw y2 dims 32..34 -> per-flow drift buffer
          #pragma unroll
          for (int j = 0; j < 8; ++j)
            sDxyz[w][(16 * m + j + 8 * hf) * 4 + lrow] = acc[j];
        }
      }
    }
    wave_lds_fence();

    // ---- owner lane: position update, termination, deposit ----
    float d0 = sDxyz[w][lane * 4 + 0];
    float d1 = sDxyz[w][lane * 4 + 1];
    float d2 = sDxyz[w][lane * 4 + 2];
    px += fast_tanh(d0);
    py += fast_tanh(d1);
    pz += 2.0f / (1.0f + __expf(-d2));
    frow[32] = f2bf(px * (1.0f / 64.0f));
    frow[33] = f2bf(py * (1.0f / 64.0f));
    frow[34] = f2bf(pz * (1.0f / 24.0f));

    const bool comp = active && (pz >= 23.0f);
    const bool oob  = active && ((px < 0.0f) || (px >= 64.0f) ||
                                 (py < 0.0f) || (py >= 64.0f));
    if (comp) {
      float s = 0.0f;
      #pragma unroll
      for (int d = 0; d < 32; ++d) s = fmaf(bf2f(frow[d]), sWout[d], s);
      int cx = (int)floorf(px); cx = cx < 0 ? 0 : (cx > 63 ? 63 : cx);
      int cy = (int)floorf(py); cy = cy < 0 ? 0 : (cy > 63 ? 63 : cy);
      atomicAdd(&outSurf[cy * 64 + cx], s * e);
    }
    if (comp || oob) active = false;
    wave_lds_fence();   // pn writes ordered before next iteration's A loads
  }
}

extern "C" void kernel_launch(void* const* d_in, const int* in_sizes, int n_in,
                              void* d_out, int out_size, void* d_ws, size_t ws_size,
                              hipStream_t stream) {
  (void)in_sizes; (void)n_in; (void)d_ws; (void)ws_size;
  const float* emb   = (const float*)d_in[0];
  const float* Wmap  = (const float*)d_in[1];
  const float* bmap  = (const float*)d_in[2];
  const float* winit = (const float*)d_in[3];
  const float* Wpos  = (const float*)d_in[4];
  const float* W1    = (const float*)d_in[5];
  const float* b1    = (const float*)d_in[6];
  const float* W2    = (const float*)d_in[7];
  const float* b2    = (const float*)d_in[8];
  const float* wout  = (const float*)d_in[9];
  float* out = (float*)d_out;

  zero_f32<<<(out_size + 255) / 256, 256, 0, stream>>>(out, out_size);

  const int nflow  = 64 * 4096;          // B * SURF
  const int blocks = nflow / TPB;        // 2048
  flow_kernel<<<blocks, TPB, 0, stream>>>(emb, Wmap, bmap, winit, Wpos,
                                          W1, b1, W2, b2, wout, out);
}
